// Seq2Seq_32384053412314
// MI455X (gfx1250) — matrix operand order repaired and checked by reading, not compiled, not run on hardware
//
#include <hip/hip_runtime.h>
#include <hip/hip_bf16.h>
#include <math.h>

// ---------------- problem constants ----------------
#define B_   256
#define T_   128
#define E_   256
#define H_   256
#define NT_  128
#define G3H  (3 * H_)        // 768
#define PAD_ID 0
#define O_ID   1

typedef __attribute__((ext_vector_type(16))) __bf16 v16bf;
typedef __attribute__((ext_vector_type(8)))  __bf16 v8bf;
typedef __attribute__((ext_vector_type(8)))  float  v8f;

// ---------------- workspace layout (bytes) ----------------
static const size_t SZW        = (size_t)G3H * E_ * 2;        // 393216
static const size_t OFF_WEIH   = 0;
static const size_t OFF_WEHH   = SZW;                          //  393216
static const size_t OFF_WDIH   = 2 * SZW;                      //  786432
static const size_t OFF_WDHH   = 3 * SZW;                      // 1179648
static const size_t OFF_WOUT   = 4 * SZW;                      // 1572864 (NT*H*2 = 65536)
static const size_t OFF_H      = 1703936;                      // B*H f32   = 262144
static const size_t OFF_HBF    = 1966080;                      // B*H bf16  = 131072
static const size_t OFF_GH     = 2097152;                      // B*3H f32  = 786432
static const size_t OFF_NLL    = 2883584;                      // T f32
static const size_t OFF_CNT    = 2884096;                      // T f32
static const size_t OFF_X      = 4194304;                      // T*B*E bf16 = 16 MB
static const size_t OFF_GI     = 33554432;                     // T*B*3H f32 = 96 MB
static const size_t OFF_LOGITS = OFF_GI;                       // reused after decoder loop
static const size_t OFF_HDEC   = 142606336;                    // T*B*H bf16 = 16 MB

// ---------------- utility kernels ----------------
__global__ void k_zero_b32(unsigned* __restrict__ p, int n) {
    int i = blockIdx.x * blockDim.x + threadIdx.x;
    if (i < n) p[i] = 0u;
}

__global__ void k_f32_to_bf16(const float* __restrict__ s, __bf16* __restrict__ d, int n) {
    int i = blockIdx.x * blockDim.x + threadIdx.x;
    if (i < n) d[i] = (__bf16)s[i];
}

// encoder input gather: X[(t*B+b)*E + e] = bf16(word_embed[input_ids[b][t]][e])
__global__ void k_gather_enc(const int* __restrict__ ids, const float* __restrict__ emb,
                             __bf16* __restrict__ X) {
    size_t i = (size_t)blockIdx.x * blockDim.x + threadIdx.x;   // T*B*E
    int e = (int)(i & (E_ - 1));
    size_t r = i >> 8;                                          // / E_
    int b = (int)(r & (B_ - 1));
    int t = (int)(r >> 8);                                      // / B_
    int tok = ids[b * T_ + t];
    X[i] = (__bf16)emb[(size_t)tok * E_ + e];
}

// decoder input gather: dec_in[t][b] = t==0 ? O_ID : tag[b][t-1]; x = relu(dec_embed[tok])
__global__ void k_gather_dec(const int* __restrict__ tags, const float* __restrict__ emb,
                             __bf16* __restrict__ X) {
    size_t i = (size_t)blockIdx.x * blockDim.x + threadIdx.x;   // T*B*H
    int e = (int)(i & (H_ - 1));
    size_t r = i >> 8;
    int b = (int)(r & (B_ - 1));
    int t = (int)(r >> 8);
    int tok = (t == 0) ? O_ID : tags[b * T_ + (t - 1)];
    float v = emb[(size_t)tok * H_ + e];
    X[i] = (__bf16)(v > 0.f ? v : 0.f);
}

// ------- 2x2 register-blocked WMMA GEMM: C[M,N] = A[M,K] * Bm[N,K]^T (+ bias) -------
// A  : M x K row-major bf16
// Bm : N x K row-major bf16 (PyTorch weight layout W[out,in])
// One wave computes a 32x32 macro-tile = four 16x16 WMMA tiles, sharing 2 A-frags
// and 2 B-frags per K-chunk: 8 b128 loads per 4 v_wmma (vs 4 loads per wmma before).
__global__ void k_wmma_gemm(const __bf16* __restrict__ A, const __bf16* __restrict__ Bm,
                            const float* __restrict__ bias, float* __restrict__ C,
                            int M, int N, int K) {
    const int lane = threadIdx.x & 31;
    const int wave = threadIdx.x >> 5;
    const int tiles_n = N >> 5;
    const int tile = blockIdx.x * (blockDim.x >> 5) + wave;
    if (tile >= (M >> 5) * tiles_n) return;            // wave-uniform: EXEC stays all-ones
    const int tm = (tile / tiles_n) << 5;
    const int tn = (tile % tiles_n) << 5;
    const int lo = lane & 15;
    const int hi = (lane >> 4) & 1;

    // A operand (16x32 bf16): lane lo = row M, hi selects K+16 half; contiguous 16 halves.
    const __bf16* a0 = A + (size_t)(tm + lo) * K + hi * 16;
    const __bf16* a1 = a0 + (size_t)16 * K;
    // B operand (32x16 bf16): lane lo = col N, hi selects K+16 half; contiguous 16 halves.
    const __bf16* b0 = Bm + (size_t)(tn + lo) * K + hi * 16;
    const __bf16* b1 = b0 + (size_t)16 * K;

    v8f acc00 = {}, acc01 = {}, acc10 = {}, acc11 = {};
    union Frag { v16bf v; v8bf h[2]; };

#pragma unroll 2
    for (int kb = 0; kb < K; kb += 32) {
        if (kb + 32 < K) {
            __builtin_prefetch(a0 + kb + 32, 0, 3);
            __builtin_prefetch(a1 + kb + 32, 0, 3);
            __builtin_prefetch(b0 + kb + 32, 0, 3);
            __builtin_prefetch(b1 + kb + 32, 0, 3);
        }
        Frag fa0, fa1, fb0, fb1;
        fa0.h[0] = *reinterpret_cast<const v8bf*>(a0 + kb);
        fa0.h[1] = *reinterpret_cast<const v8bf*>(a0 + kb + 8);
        fa1.h[0] = *reinterpret_cast<const v8bf*>(a1 + kb);
        fa1.h[1] = *reinterpret_cast<const v8bf*>(a1 + kb + 8);
        fb0.v    = *reinterpret_cast<const v16bf*>(b0 + kb);
        fb1.v    = *reinterpret_cast<const v16bf*>(b1 + kb);
        acc00 = __builtin_amdgcn_wmma_f32_16x16x32_bf16(
                    false, fa0.v, false, fb0.v, (short)0, acc00, false, false);
        acc01 = __builtin_amdgcn_wmma_f32_16x16x32_bf16(
                    false, fa0.v, false, fb1.v, (short)0, acc01, false, false);
        acc10 = __builtin_amdgcn_wmma_f32_16x16x32_bf16(
                    false, fa1.v, false, fb0.v, (short)0, acc10, false, false);
        acc11 = __builtin_amdgcn_wmma_f32_16x16x32_bf16(
                    false, fa1.v, false, fb1.v, (short)0, acc11, false, false);
    }

    const float bn0 = bias ? bias[tn + lo] : 0.f;
    const float bn1 = bias ? bias[tn + 16 + lo] : 0.f;
    float* crow = C + (size_t)tm * N + tn + lo;
#pragma unroll
    for (int r = 0; r < 8; ++r) {
        int m = r + hi * 8;                 // C/D layout: VGPR r -> M, hi -> +8
        crow[(size_t)m * N]             = acc00[r] + bn0;
        crow[(size_t)m * N + 16]        = acc01[r] + bn1;
        crow[(size_t)(m + 16) * N]      = acc10[r] + bn0;
        crow[(size_t)(m + 16) * N + 16] = acc11[r] + bn1;
    }
}

// ---------------- GRU pointwise gate fusion ----------------
// gates ordered (r, z, n); gi already has b_ih, gh already has b_hh.
__global__ void k_gru_pointwise(const float* __restrict__ GI, const float* __restrict__ GH,
                                float* __restrict__ h, __bf16* __restrict__ hbf,
                                __bf16* __restrict__ hist, int t) {
    int i = blockIdx.x * blockDim.x + threadIdx.x;  // B*H
    int b = i >> 8;                                  // / H_
    int j = i & (H_ - 1);
    const float* gi = GI + ((size_t)t * B_ + b) * G3H;
    const float* gh = GH + (size_t)b * G3H;
    float r = 1.f / (1.f + __expf(-(gi[j] + gh[j])));
    float z = 1.f / (1.f + __expf(-(gi[H_ + j] + gh[H_ + j])));
    float n = tanhf(gi[2 * H_ + j] + r * gh[2 * H_ + j]);
    float hnew = (1.f - z) * n + z * h[i];
    h[i] = hnew;
    __bf16 hb = (__bf16)hnew;
    hbf[i] = hb;
    if (hist) hist[((size_t)t * B_ + b) * H_ + j] = hb;
}

// ---------------- fused log-softmax (NT=128) + NLL accumulation ----------------
// one wave32 per (t,b) row; writes logp transposed to (B,T,NT) output.
__global__ void k_logsoftmax_nll(const float* __restrict__ logits, const int* __restrict__ tags,
                                 float* __restrict__ out, float* __restrict__ nll_sum,
                                 float* __restrict__ cnt) {
    int row  = blockIdx.x * (blockDim.x >> 5) + (threadIdx.x >> 5);   // over T*B
    int lane = threadIdx.x & 31;
    if (row >= T_ * B_) return;
    int b = row & (B_ - 1);
    int t = row >> 8;
    const float* lr = logits + (size_t)row * NT_;
    float v[4], m = -1e30f;
#pragma unroll
    for (int q = 0; q < 4; ++q) { v[q] = lr[lane + q * 32]; m = fmaxf(m, v[q]); }
#pragma unroll
    for (int o = 16; o > 0; o >>= 1) m = fmaxf(m, __shfl_xor(m, o, 32));
    float s = 0.f;
#pragma unroll
    for (int q = 0; q < 4; ++q) s += __expf(v[q] - m);
#pragma unroll
    for (int o = 16; o > 0; o >>= 1) s += __shfl_xor(s, o, 32);
    float lse = m + __logf(s);
    float* orow = out + ((size_t)b * T_ + t) * NT_;
#pragma unroll
    for (int q = 0; q < 4; ++q) orow[lane + q * 32] = v[q] - lse;
    if (lane == 0) {
        int tgt = tags[b * T_ + t];                 // tag_T[t][b]
        if (tgt != PAD_ID) {
            atomicAdd(&nll_sum[t], -(lr[tgt] - lse));
            atomicAdd(&cnt[t], 1.f);
        }
    }
}

__global__ void k_final_loss(const float* __restrict__ nll_sum, const float* __restrict__ cnt,
                             float* __restrict__ out_loss) {
    if (threadIdx.x == 0 && blockIdx.x == 0) {
        float acc = 0.f;
        for (int t = 0; t < T_; ++t)
            acc += (cnt[t] > 0.f) ? (nll_sum[t] / cnt[t]) : 0.f;
        *out_loss = acc;
    }
}

// ---------------- host orchestration ----------------
extern "C" void kernel_launch(void* const* d_in, const int* in_sizes, int n_in,
                              void* d_out, int out_size, void* d_ws, size_t ws_size,
                              hipStream_t stream) {
    const int*   input_ids  = (const int*)  d_in[0];
    const int*   tag_ids    = (const int*)  d_in[1];
    const float* word_embed = (const float*)d_in[2];
    const float* enc_W_ih   = (const float*)d_in[3];
    const float* enc_W_hh   = (const float*)d_in[4];
    const float* enc_b_ih   = (const float*)d_in[5];
    const float* enc_b_hh   = (const float*)d_in[6];
    const float* dec_embed  = (const float*)d_in[7];
    const float* dec_W_ih   = (const float*)d_in[8];
    const float* dec_W_hh   = (const float*)d_in[9];
    const float* dec_b_ih   = (const float*)d_in[10];
    const float* dec_b_hh   = (const float*)d_in[11];
    const float* out_W      = (const float*)d_in[12];
    const float* out_b      = (const float*)d_in[13];

    char* ws = (char*)d_ws;
    __bf16* wEih = (__bf16*)(ws + OFF_WEIH);
    __bf16* wEhh = (__bf16*)(ws + OFF_WEHH);
    __bf16* wDih = (__bf16*)(ws + OFF_WDIH);
    __bf16* wDhh = (__bf16*)(ws + OFF_WDHH);
    __bf16* wOut = (__bf16*)(ws + OFF_WOUT);
    float*  h    = (float*) (ws + OFF_H);
    __bf16* hbf  = (__bf16*)(ws + OFF_HBF);
    float*  gh   = (float*) (ws + OFF_GH);
    float*  nll  = (float*) (ws + OFF_NLL);
    float*  cnt  = (float*) (ws + OFF_CNT);
    __bf16* X    = (__bf16*)(ws + OFF_X);
    float*  GI   = (float*) (ws + OFF_GI);
    float*  LG   = (float*) (ws + OFF_LOGITS);
    __bf16* Hdec = (__bf16*)(ws + OFF_HDEC);
    float*  out  = (float*)d_out;

    // --- weight f32 -> bf16 (off the recurrence path) ---
    k_f32_to_bf16<<<(G3H * E_) / 256, 256, 0, stream>>>(enc_W_ih, wEih, G3H * E_);
    k_f32_to_bf16<<<(G3H * H_) / 256, 256, 0, stream>>>(enc_W_hh, wEhh, G3H * H_);
    k_f32_to_bf16<<<(G3H * H_) / 256, 256, 0, stream>>>(dec_W_ih, wDih, G3H * H_);
    k_f32_to_bf16<<<(G3H * H_) / 256, 256, 0, stream>>>(dec_W_hh, wDhh, G3H * H_);
    k_f32_to_bf16<<<(NT_ * H_) / 256, 256, 0, stream>>>(out_W,    wOut, NT_ * H_);

    // --- zero state + accumulators ---
    k_zero_b32<<<(B_ * H_) / 256, 256, 0, stream>>>((unsigned*)h, B_ * H_);
    k_zero_b32<<<(B_ * H_ / 2) / 256, 256, 0, stream>>>((unsigned*)hbf, B_ * H_ / 2);
    k_zero_b32<<<1, 256, 0, stream>>>((unsigned*)nll, 2 * T_);   // nll + cnt contiguous

    // 32x32 macro-tiles per wave
    const int tilesGI = (T_ * B_ / 32) * (G3H / 32);   // 1024*24 = 24576
    const int tilesGH = (B_ / 32) * (G3H / 32);        //    8*24 = 192
    const int tilesLG = (T_ * B_ / 32) * (NT_ / 32);   //  1024*4 = 4096

    // ===== encoder =====
    k_gather_enc<<<(T_ * B_ * E_) / 256, 256, 0, stream>>>(input_ids, word_embed, X);
    k_wmma_gemm<<<tilesGI / 8, 256, 0, stream>>>(X, wEih, enc_b_ih, GI, T_ * B_, G3H, E_);
    for (int t = 0; t < T_; ++t) {
        k_wmma_gemm<<<tilesGH / 8, 256, 0, stream>>>(hbf, wEhh, enc_b_hh, gh, B_, G3H, H_);
        k_gru_pointwise<<<(B_ * H_) / 256, 256, 0, stream>>>(GI, gh, h, hbf, nullptr, t);
    }

    // ===== decoder (teacher forcing; inputs known upfront) =====
    k_gather_dec<<<(T_ * B_ * H_) / 256, 256, 0, stream>>>(tag_ids, dec_embed, X);
    k_wmma_gemm<<<tilesGI / 8, 256, 0, stream>>>(X, wDih, dec_b_ih, GI, T_ * B_, G3H, H_);
    for (int t = 0; t < T_; ++t) {
        k_wmma_gemm<<<tilesGH / 8, 256, 0, stream>>>(hbf, wDhh, dec_b_hh, gh, B_, G3H, H_);
        k_gru_pointwise<<<(B_ * H_) / 256, 256, 0, stream>>>(GI, gh, h, hbf, Hdec, t);
    }

    // ===== batched output projection + log-softmax + loss =====
    k_wmma_gemm<<<tilesLG / 8, 256, 0, stream>>>(Hdec, wOut, out_b, LG, T_ * B_, NT_, H_);
    k_logsoftmax_nll<<<(T_ * B_) / 8, 256, 0, stream>>>(LG, tag_ids, out, nll, cnt);
    k_final_loss<<<1, 32, 0, stream>>>(nll, cnt, out + (size_t)B_ * T_ * NT_);
}
